// RNN_MultiRegional_NoConstraintThal_29085518528928
// MI455X (gfx1250) — matrix operand order repaired and checked by reading, not compiled, not run on hardware
//
#include <hip/hip_runtime.h>
#include <hip/hip_bf16.h>

typedef __bf16 bf16_t;
typedef bf16_t v16bf __attribute__((ext_vector_type(16)));
typedef bf16_t v8bf  __attribute__((ext_vector_type(8)));
typedef float  v8f   __attribute__((ext_vector_type(8)));

#define H3   3072
#define NB   64
#define NT   1000
#define TC   0.01f
#define PH_MULT 1.41421356e-4f   /* sqrt(2*0.01*0.001^2) */
#define PI_MULT 2.1213203e-2f    /* sqrt(2*0.01*0.15^2)  */

// d_out flat layout (floats): out | hn_last | rnn_out | x_last | x_out
#define OFF_OUT   ((size_t)0)
#define OFF_HN    ((size_t)64000)
#define OFF_RNN   ((size_t)260608)
#define OFF_XLAST ((size_t)196868608)
#define OFF_XOUT  ((size_t)197065216)

#define NBLOCKS 48

// ---------------- build W_rec (bf16, row-major [3072][3072]) ----------------
__global__ void k_build_w(const float* __restrict__ s2s, const float* __restrict__ t2a,
                          const float* __restrict__ t2s, const float* __restrict__ a2a,
                          const float* __restrict__ a2s, const float* __restrict__ s2t,
                          bf16_t* __restrict__ W) {
  int tid = blockIdx.x * 256 + threadIdx.x;
  if (tid >= H3 * H3) return;
  int r = tid / H3, c = tid - r * H3;
  int rb = r >> 10, cb = c >> 10, rl = r & 1023, cl = c & 1023;
  const float* src = nullptr;
  if (rb == 0)      src = (cb == 0) ? s2s : (cb == 1) ? t2s : a2s;  // str rows
  else if (rb == 1) { if (cb == 0) src = s2t; }                     // thal rows
  else              { if (cb == 1) src = t2a; else if (cb == 2) src = a2a; } // alm rows
  float v = src ? src[rl * 1024 + cl] : 0.0f;
  W[tid] = (bf16_t)v;
}

// -------- init: step counters, s[t][b] = inp[b,t] + pi_mult*pinp[t], h0 -----
__global__ void k_init(const float* __restrict__ inp, const float* __restrict__ pinp,
                       const float* __restrict__ hn_in,
                       float* __restrict__ s_arr, bf16_t* __restrict__ hbuf0,
                       int* __restrict__ counters) {
  int tid = blockIdx.x * 256 + threadIdx.x;
  if (tid < NT) counters[tid] = 0;
  if (tid < NB * NT) {
    int b = tid & 63, t = tid >> 6;
    s_arr[tid] = inp[b * NT + t] + PI_MULT * pinp[t];
  }
  if (tid < NB * H3) hbuf0[tid] = (bf16_t)hn_in[tid];
}

// ------------------------- persistent RNN kernel ----------------------------
// 48 blocks x 256 threads (8 waves). Wave handles 32(M) x 16(N) output tile.
// Block bk covers columns [64*bk, 64*bk+64); wave pair (w>>1) picks N-tile,
// (w&1) picks M half (rows 0-31 or 32-63).
__global__ __launch_bounds__(256) void k_rnn(
    const bf16_t* __restrict__ W, bf16_t* __restrict__ hbuf,
    const float* __restrict__ s_arr, int* __restrict__ counters,
    const float* __restrict__ x_in, const float* __restrict__ inhib,
    const float* __restrict__ inp_w, const float* __restrict__ p_hid,
    float* __restrict__ d_out) {
  const int wave = threadIdx.x >> 5;
  const int lane = threadIdx.x & 31;
  const int hi   = lane >> 4;
  const int l15  = lane & 15;
  const int m0   = (wave & 1) * 32;                       // row base (2 tiles)
  const int j0   = ((blockIdx.x << 2) + (wave >> 1)) << 4; // column tile base
  const int j    = j0 + l15;                               // this lane's column

  // masked input weight: strthal_mask = 1 only for 512 <= j < 1024
  const float wj = (j >= 512 && j < 1024) ? inp_w[j] : 0.0f;

  // persistent per-element state (x) and hoisted inhib_stim
  float xr[2][8], ih[2][8];
#pragma unroll
  for (int tt = 0; tt < 2; ++tt)
#pragma unroll
    for (int i = 0; i < 8; ++i) {
      int b = m0 + tt * 16 + hi * 8 + i;
      xr[tt][i] = x_in[b * H3 + j];
      ih[tt][i] = inhib[b * H3 + j];
    }

  const bf16_t* Bp = W + (size_t)j * H3 + hi * 16;  // B tile: contiguous 16 bf16/lane

  for (int t = 0; t < NT; ++t) {
    if (t > 0) {
      if (threadIdx.x == 0) {
        while (__hip_atomic_load(&counters[t - 1], __ATOMIC_RELAXED,
                                 __HIP_MEMORY_SCOPE_AGENT) < NBLOCKS)
          __builtin_amdgcn_s_sleep(1);
      }
      __syncthreads();
      __builtin_amdgcn_fence(__ATOMIC_ACQUIRE, "agent");
    }

    const bf16_t* Hb = hbuf + (size_t)(t & 1) * (NB * H3);
    bf16_t*       Hw = hbuf + (size_t)((t + 1) & 1) * (NB * H3);
    const bf16_t* Ar0 = Hb + (size_t)(m0 + l15) * H3 + hi * 16;
    const bf16_t* Ar1 = Ar0 + (size_t)16 * H3;

    v8f c0 = {0.f,0.f,0.f,0.f,0.f,0.f,0.f,0.f};
    v8f c1 = {0.f,0.f,0.f,0.f,0.f,0.f,0.f,0.f};

#pragma unroll 4
    for (int k = 0; k < H3; k += 32) {
      v8bf a0l = *(const v8bf*)(Ar0 + k);
      v8bf a0h = *(const v8bf*)(Ar0 + k + 8);
      v8bf a1l = *(const v8bf*)(Ar1 + k);
      v8bf a1h = *(const v8bf*)(Ar1 + k + 8);
      v16bf bm = *(const v16bf*)(Bp + k);
      v16bf a0 = __builtin_shufflevector(a0l, a0h, 0,1,2,3,4,5,6,7,8,9,10,11,12,13,14,15);
      v16bf a1 = __builtin_shufflevector(a1l, a1h, 0,1,2,3,4,5,6,7,8,9,10,11,12,13,14,15);
      c0 = __builtin_amdgcn_wmma_f32_16x16x32_bf16(false, a0, false, bm, (short)0, c0, false, false);
      c1 = __builtin_amdgcn_wmma_f32_16x16x32_bf16(false, a1, false, bm, (short)0, c1, false, false);
    }

    const float ph_t = PH_MULT * p_hid[t];
#pragma unroll
    for (int tt = 0; tt < 2; ++tt) {
      v8f& cc = tt ? c1 : c0;
#pragma unroll
      for (int i = 0; i < 8; ++i) {
        int b = m0 + tt * 16 + hi * 8 + i;
        float sv = s_arr[t * 64 + b];
        float xo = xr[tt][i];
        float xn = xo + TC * (-xo + cc[i] + sv * wj + ih[tt][i]) + ph_t;
        float hv = fmaxf(xn, 0.0f);
        xr[tt][i] = xn;
        size_t bo = (size_t)b * (NT * H3) + (size_t)t * H3 + (size_t)j;
        d_out[OFF_XOUT + bo] = xn;
        d_out[OFF_RNN + bo] = hv;
        Hw[b * H3 + j] = (bf16_t)hv;
        if (t == NT - 1) {
          d_out[OFF_HN + (size_t)b * H3 + j]    = hv;
          d_out[OFF_XLAST + (size_t)b * H3 + j] = xn;
        }
      }
    }

    __builtin_amdgcn_fence(__ATOMIC_RELEASE, "agent");
    __syncthreads();
    if (threadIdx.x == 0)
      __hip_atomic_fetch_add(&counters[t], 1, __ATOMIC_RELAXED, __HIP_MEMORY_SCOPE_AGENT);
  }
}

// ---------- out[b,t] = sum_{j>=2048} h[b,t,j]*fc1_w[j] + fc1_b --------------
__global__ void k_out(float* __restrict__ d_out, const float* __restrict__ fc1w,
                      const float* __restrict__ fc1b) {
  int gw = (blockIdx.x * 256 + threadIdx.x) >> 5;
  int lane = threadIdx.x & 31;
  if (gw >= NB * NT) return;
  int b = gw / NT, t = gw - b * NT;
  const float* hp = d_out + OFF_RNN + (size_t)b * (NT * H3) + (size_t)t * H3 + 2048;
  const float* wp = fc1w + 2048;
  float acc = 0.0f;
#pragma unroll
  for (int i = 0; i < 32; ++i) acc += hp[lane + 32 * i] * wp[lane + 32 * i];
#pragma unroll
  for (int off = 16; off; off >>= 1) acc += __shfl_xor(acc, off, 32);
  if (lane == 0) d_out[(size_t)b * NT + t] = acc + fc1b[0];
}

extern "C" void kernel_launch(void* const* d_in, const int* in_sizes, int n_in,
                              void* d_out, int out_size, void* d_ws, size_t ws_size,
                              hipStream_t stream) {
  const float* inp   = (const float*)d_in[0];
  const float* hn    = (const float*)d_in[1];
  const float* x0    = (const float*)d_in[2];
  const float* inhib = (const float*)d_in[3];
  const float* phid  = (const float*)d_in[4];
  const float* pinp  = (const float*)d_in[5];
  const float* s2s   = (const float*)d_in[6];
  const float* t2a   = (const float*)d_in[7];
  const float* t2s   = (const float*)d_in[8];
  const float* a2a   = (const float*)d_in[9];
  const float* a2s   = (const float*)d_in[10];
  const float* s2t   = (const float*)d_in[11];
  const float* inpw  = (const float*)d_in[12];
  const float* fc1w  = (const float*)d_in[13];
  const float* fc1b  = (const float*)d_in[14];
  float* out = (float*)d_out;

  char* ws = (char*)d_ws;
  bf16_t* W       = (bf16_t*)ws;                                  // 18,874,368 B
  bf16_t* hb      = (bf16_t*)(ws + 18874368);                     //    786,432 B
  float*  s_arr   = (float*)(ws + 18874368 + 786432);             //    256,000 B
  int*    counters= (int*)(ws + 18874368 + 786432 + 256000);      //      4,000 B

  k_build_w<<<(H3 * H3 + 255) / 256, 256, 0, stream>>>(s2s, t2a, t2s, a2a, a2s, s2t, W);
  k_init<<<(NB * H3 + 255) / 256, 256, 0, stream>>>(inp, pinp, hn, s_arr, hb, counters);
  k_rnn<<<NBLOCKS, 256, 0, stream>>>(W, hb, s_arr, counters, x0, inhib, inpw, phid, out);
  k_out<<<(NB * NT) / 8, 256, 0, stream>>>(out, fc1w, fc1b);
}
